// Attention_61469571940467
// MI455X (gfx1250) — matrix operand order repaired and checked by reading, not compiled, not run on hardware
//
#include <hip/hip_runtime.h>
#include <cstdint>
#include <cstddef>

// ---------------------------------------------------------------------------
// CDNA5 (gfx1250) attention:
//   out = softmax((x Wq^T + bq)(x Wk^T + bk)^T / 32) (x Wv^T + bv)
// bf16 WMMA GEMMs fed by the Tensor Data Mover (tensor_load_to_lds, TENSORcnt,
// double-buffered LDS tiles), L2-resident bf16 score matrix (134 MB < 192 MB L2),
// in-place LDS row softmax. All intermediates carved from d_ws.
// ---------------------------------------------------------------------------

typedef __attribute__((ext_vector_type(16))) __bf16          v16bf;
typedef __attribute__((ext_vector_type(8)))  __bf16          v8bf;
typedef __attribute__((ext_vector_type(8)))  float           v8f;
typedef __attribute__((ext_vector_type(8)))  unsigned short  v8us;
typedef __attribute__((ext_vector_type(4)))  unsigned int    v4u;
typedef __attribute__((ext_vector_type(8)))  int             v8i;
typedef __attribute__((ext_vector_type(4)))  int             v4i;

__device__ __forceinline__ unsigned short f32_to_bf16(float f) {
  unsigned u = __float_as_uint(f);
  u += 0x7FFFu + ((u >> 16) & 1u);          // round-to-nearest-even
  return (unsigned short)(u >> 16);
}
__device__ __forceinline__ float bf16_to_f32(unsigned short h) {
  return __uint_as_float(((unsigned)h) << 16);
}

union ABFrag { v16bf v; v8bf h[2]; };

// ---------------------------------------------------------------------------
// TDM: DMA a [rows x tileK] bf16 tile (row stride = strideElems) from global
// into LDS at byte offset ldsOff, padded +8 elements (4 DWORDs) per 64-element
// row so strided fragment reads avoid LDS bank conflicts.
// D# layout per CDNA5 ISA ch.8 (group0: count/lds/global/type, group1: dims).
// ---------------------------------------------------------------------------
__device__ __forceinline__ void tdm_load_tile(unsigned ldsOff,
                                              const unsigned short* gptr,
                                              unsigned rows, unsigned tileK,
                                              unsigned strideElems) {
#if defined(__HIP_DEVICE_COMPILE__)
  unsigned long long ga = (unsigned long long)(uintptr_t)gptr;
  v4u g0;
  g0[0] = 1u;                                             // count=1, user mode
  g0[1] = ldsOff;                                         // lds_addr (bytes)
  g0[2] = (unsigned)(ga & 0xffffffffu);                   // global_addr[31:0]
  g0[3] = (unsigned)((ga >> 32) & 0x1ffffffu)             // global_addr[56:32]
          | (2u << 30);                                   // type = 2 ("image")
  const unsigned td0 = strideElems;                       // tensor_dim0 (>= tile)
  const unsigned td1 = rows;                              // tensor_dim1
  const unsigned long long s0 = strideElems;              // tensor_dim0_stride
  v8i g1;
  g1[0] = (int)((1u << 16)      // data_size = 2 bytes (bf16)
              | (1u << 20)      // pad_enable
              | (4u << 22)      // pad_interval code 4 -> every 32 DWORDs (64 elem)
              | (3u << 25));    // pad_amount  code 3 -> 4 DWORDs (8 elem)
  g1[1] = (int)((td0 & 0xffffu) << 16);                         // td0[15:0]
  g1[2] = (int)(((td0 >> 16) & 0xffffu) | ((td1 & 0xffffu) << 16));
  g1[3] = (int)(((td1 >> 16) & 0xffffu) | ((tileK & 0xffffu) << 16)); // tile_dim0
  g1[4] = (int)(rows & 0xffffu);                                // tile_dim1, dim2=0
  g1[5] = (int)(s0 & 0xffffffffu);
  g1[6] = (int)((s0 >> 32) & 0xffffu);                          // stride1 = 0
  g1[7] = 0;
  v4i z4 = {0, 0, 0, 0};
#if __clang_major__ >= 23
  v8i z8 = {0, 0, 0, 0, 0, 0, 0, 0};
  __builtin_amdgcn_tensor_load_to_lds(g0, g1, z4, z4, z8, 0);
#else
  __builtin_amdgcn_tensor_load_to_lds(g0, g1, z4, z4, 0);
#endif
#else
  (void)ldsOff; (void)gptr; (void)rows; (void)tileK; (void)strideElems;
#endif
}

// s_wait_tensorcnt requires a compile-time-constant SIMM16 -> template param.
template <int CNT>
__device__ __forceinline__ void wait_tensorcnt_le() {
#if defined(__HIP_DEVICE_COMPILE__)
  __builtin_amdgcn_s_wait_tensorcnt(CNT);
#endif
}

// ---------------------------------------------------------------------------
// fp32 -> bf16 conversion (grid-stride)
// ---------------------------------------------------------------------------
__global__ __launch_bounds__(256) void cvt_f32_bf16(const float* __restrict__ s,
                                                    unsigned short* __restrict__ d,
                                                    int n) {
  int i = blockIdx.x * 256 + threadIdx.x;
  int stride = gridDim.x * 256;
  for (; i < n; i += stride) d[i] = f32_to_bf16(s[i]);
}

// ---------------------------------------------------------------------------
// C[M,N] = alpha * (A[MxK] @ B[NxK]^T) + bias[N]
// A, B bf16 row-major along K. Block tile 128x128 (8 waves, 4x2 of 32x64).
// K-chunks of 64 are DMA'd by the TDM into double-buffered padded LDS tiles;
// waves read WMMA fragments from LDS (ds_load_b128) and issue
// v_wmma_f32_16x16x32_bf16.
// OUT_MODE: 0 = f32 row-major, 1 = bf16 row-major, 2 = bf16 transposed C[n][m]
// M, N multiples of 128; K multiple of 64.
// ---------------------------------------------------------------------------
template <int OUT_MODE>
__global__ __launch_bounds__(256) void wmma_gemm_bt(
    const unsigned short* __restrict__ A,
    const unsigned short* __restrict__ B,
    const float* __restrict__ bias,
    void* __restrict__ Cout,
    int M, int N, int K, int ldc, float alpha) {
  (void)M; (void)N;
  constexpr int KC    = 64;         // k-chunk per TDM tile
  constexpr int LDSK  = KC + 8;     // padded LDS row stride (elements)
  constexpr int TILE  = 128 * LDSK; // elements per tile buffer
  constexpr int TILEB = TILE * 2;   // bytes per tile buffer

  extern __shared__ unsigned short smem[];  // 4 * TILE elements = 73728 B
  // Layout: [ A buf0 | A buf1 | B buf0 | B buf1 ]  (no pointer arrays: keep all
  // LDS addressing as runtime arithmetic to avoid addrspacecast static inits)

  const int lane   = threadIdx.x & 31;
  const int wave   = threadIdx.x >> 5;
  const int wy     = wave & 3;              // 4 waves along M
  const int wx     = wave >> 2;             // 2 waves along N
  const int laneM  = lane & 15;
  const int laneHi = lane >> 4;

  const unsigned short* Ablk = A + (size_t)(blockIdx.y * 128) * K;
  const unsigned short* Bblk = B + (size_t)(blockIdx.x * 128) * K;
  const bool tdmWave = (threadIdx.x < 32);

  // LDS byte offsets of the four buffers (runtime cast, not a static init).
  const unsigned ldsBase = (unsigned)(uintptr_t)(void*)smem;

  // Preload k-chunk 0 into buffer 0.
  if (tdmWave) {
    tdm_load_tile(ldsBase, Ablk, 128, KC, (unsigned)K);
    tdm_load_tile(ldsBase + 2u * TILEB, Bblk, 128, KC, (unsigned)K);
  }

  v8f acc[2][4] = {};

  // Per-lane LDS element offsets (A and B: two 16B runs each, 32B apart).
  const int aOff0 = (wy * 32 + laneM) * LDSK + laneHi * 8;   // + mt*16*LDSK
  const int bOff0 = (wx * 64 + laneM) * LDSK + laneHi * 8;   // + nt*16*LDSK

  const int nk = K / KC;
  for (int ki = 0; ki < nk; ++ki) {
    __syncthreads();  // everyone done reading the buffer TDM is about to reuse
    if (tdmWave) {
      if (ki + 1 < nk) {
        const unsigned nb = (unsigned)((ki + 1) & 1);
        tdm_load_tile(ldsBase + nb * TILEB,
                      Ablk + (size_t)(ki + 1) * KC, 128, KC, (unsigned)K);
        tdm_load_tile(ldsBase + 2u * TILEB + nb * TILEB,
                      Bblk + (size_t)(ki + 1) * KC, 128, KC, (unsigned)K);
        wait_tensorcnt_le<2>();  // newest pair may fly; current buffer complete
      } else {
        wait_tensorcnt_le<0>();
      }
    }
    __syncthreads();            // current buffer visible to all waves

    const unsigned short* sA = smem + (ki & 1) * TILE;
    const unsigned short* sB = smem + 2 * TILE + (ki & 1) * TILE;

#pragma unroll
    for (int ks = 0; ks < KC; ks += 32) {
      ABFrag a[2];
#pragma unroll
      for (int mt = 0; mt < 2; ++mt) {
        const int o = aOff0 + mt * 16 * LDSK + ks;
        a[mt].h[0] = *(const v8bf*)(sA + o);
        a[mt].h[1] = *(const v8bf*)(sA + o + 16);
      }
      ABFrag b[4];
#pragma unroll
      for (int nt = 0; nt < 4; ++nt) {
        const int o = bOff0 + nt * 16 * LDSK + ks;
        b[nt].h[0] = *(const v8bf*)(sB + o);
        b[nt].h[1] = *(const v8bf*)(sB + o + 16);
      }
#pragma unroll
      for (int nt = 0; nt < 4; ++nt) {
        acc[0][nt] = __builtin_amdgcn_wmma_f32_16x16x32_bf16(
            false, a[0].v, false, b[nt].v, (short)0, acc[0][nt], false, false);
        acc[1][nt] = __builtin_amdgcn_wmma_f32_16x16x32_bf16(
            false, a[1].v, false, b[nt].v, (short)0, acc[1][nt], false, false);
      }
    }
  }

  // Epilogue. D layout: VGPR v, lane L -> (m = v + 8*(L/16), n = L%16).
  const int bm = blockIdx.y * 128 + wy * 32;
  const int bn = blockIdx.x * 128 + wx * 64;
#pragma unroll
  for (int mt = 0; mt < 2; ++mt) {
#pragma unroll
    for (int nt = 0; nt < 4; ++nt) {
      const int col  = bn + nt * 16 + laneM;
      const int rowB = bm + mt * 16 + (laneHi << 3);
      const float bv = bias ? bias[col] : 0.0f;
      if (OUT_MODE == 0) {
        float* C = (float*)Cout;
#pragma unroll
        for (int v = 0; v < 8; ++v)
          C[(size_t)(rowB + v) * ldc + col] = acc[mt][nt][v] * alpha + bv;
      } else if (OUT_MODE == 1) {
        unsigned short* C = (unsigned short*)Cout;
#pragma unroll
        for (int v = 0; v < 8; ++v)
          C[(size_t)(rowB + v) * ldc + col] =
              f32_to_bf16(acc[mt][nt][v] * alpha + bv);
      } else {  // transposed bf16: C[col][rowB+v], v contiguous -> one 16B store
        unsigned short* C = (unsigned short*)Cout;
        v8us pk;
#pragma unroll
        for (int v = 0; v < 8; ++v)
          pk[v] = f32_to_bf16(acc[mt][nt][v] * alpha + bv);
        *(v8us*)(C + (size_t)col * ldc + rowB) = pk;
      }
    }
  }
}

// ---------------------------------------------------------------------------
// In-place row softmax over bf16 [nrows x ncols], one 256-thread block per row.
// Row cached in LDS (ncols <= 8192).
// ---------------------------------------------------------------------------
__global__ __launch_bounds__(256) void softmax_rows_inplace(unsigned short* __restrict__ S,
                                                            int ncols) {
  __shared__ float buf[8192];
  __shared__ float red[256];
  unsigned short* sp = S + (size_t)blockIdx.x * ncols;
  const int t = threadIdx.x;

  float lmax = -3.4e38f;
  for (int i = t; i < ncols; i += 256) {
    float v = bf16_to_f32(sp[i]);
    buf[i] = v;
    lmax = fmaxf(lmax, v);
  }
  red[t] = lmax;
  __syncthreads();
  for (int s = 128; s > 0; s >>= 1) {
    if (t < s) red[t] = fmaxf(red[t], red[t + s]);
    __syncthreads();
  }
  const float m = red[0];
  __syncthreads();

  float lsum = 0.0f;
  for (int i = t; i < ncols; i += 256) {
    float e = __expf(buf[i] - m);
    buf[i] = e;
    lsum += e;
  }
  red[t] = lsum;
  __syncthreads();
  for (int s = 128; s > 0; s >>= 1) {
    if (t < s) red[t] += red[t + s];
    __syncthreads();
  }
  const float inv = 1.0f / red[0];
  __syncthreads();

  for (int i = t; i < ncols; i += 256)
    sp[i] = f32_to_bf16(buf[i] * inv);
}

// ---------------------------------------------------------------------------
// Host-side orchestration
// ---------------------------------------------------------------------------
extern "C" void kernel_launch(void* const* d_in, const int* in_sizes, int n_in,
                              void* d_out, int out_size, void* d_ws, size_t ws_size,
                              hipStream_t stream) {
  (void)in_sizes; (void)n_in; (void)out_size; (void)ws_size;

  constexpr int Nn = 8192;   // rows
  constexpr int E  = 1024;   // embedding
  constexpr int D  = 1024;   // q/k/v dim
  constexpr size_t GEMM_LDS = 4u * 128u * (64u + 8u) * 2u;  // 73728 B dynamic LDS

  const float* x  = (const float*)d_in[0];
  const float* Wq = (const float*)d_in[1];
  const float* bq = (const float*)d_in[2];
  const float* Wk = (const float*)d_in[3];
  const float* bk = (const float*)d_in[4];
  const float* Wv = (const float*)d_in[5];
  const float* bv = (const float*)d_in[6];
  float* out = (float*)d_out;

  // Workspace carve-out (~206 MB): bf16 x, weights, Q, K, V^T, scores.
  char* ws = (char*)d_ws;
  size_t off = 0;
  auto carve = [&](size_t bytes) -> void* {
    void* p = ws + off;
    off += (bytes + 255) & ~(size_t)255;
    return p;
  };
  unsigned short* xb  = (unsigned short*)carve((size_t)Nn * E * 2);
  unsigned short* wqb = (unsigned short*)carve((size_t)D * E * 2);
  unsigned short* wkb = (unsigned short*)carve((size_t)D * E * 2);
  unsigned short* wvb = (unsigned short*)carve((size_t)D * E * 2);
  unsigned short* qb  = (unsigned short*)carve((size_t)Nn * D * 2);   // [N,D]
  unsigned short* kb  = (unsigned short*)carve((size_t)Nn * D * 2);   // [N,D]
  unsigned short* vt  = (unsigned short*)carve((size_t)D * Nn * 2);   // [D,N] (V^T)
  unsigned short* sb  = (unsigned short*)carve((size_t)Nn * Nn * 2);  // scores

  // 1) fp32 -> bf16 conversions
  cvt_f32_bf16<<<(Nn * E + 255) / 256, 256, 0, stream>>>(x,  xb,  Nn * E);
  cvt_f32_bf16<<<(D * E + 255) / 256, 256, 0, stream>>>(Wq, wqb, D * E);
  cvt_f32_bf16<<<(D * E + 255) / 256, 256, 0, stream>>>(Wk, wkb, D * E);
  cvt_f32_bf16<<<(D * E + 255) / 256, 256, 0, stream>>>(Wv, wvb, D * E);

  // 2) Projections: Q,K row-major bf16; V stored transposed for the P@V GEMM.
  dim3 gProj(D / 128, Nn / 128);
  wmma_gemm_bt<1><<<gProj, 256, GEMM_LDS, stream>>>(xb, wqb, bq, qb, Nn, D, E, D,  1.0f);
  wmma_gemm_bt<1><<<gProj, 256, GEMM_LDS, stream>>>(xb, wkb, bk, kb, Nn, D, E, D,  1.0f);
  wmma_gemm_bt<2><<<gProj, 256, GEMM_LDS, stream>>>(xb, wvb, bv, vt, Nn, D, E, Nn, 1.0f);

  // 3) S = (Q K^T) / sqrt(D), bf16 [N,N] (L2-resident)
  dim3 gS(Nn / 128, Nn / 128);
  wmma_gemm_bt<1><<<gS, 256, GEMM_LDS, stream>>>(qb, kb, nullptr, sb, Nn, Nn, D, Nn,
                                                 1.0f / 32.0f);

  // 4) row softmax in place (P overwrites S)
  softmax_rows_inplace<<<Nn, 256, 0, stream>>>(sb, Nn);

  // 5) out = P @ V  (== P[N,N] @ (V^T)[D,N]^T), f32 output
  dim3 gO(D / 128, Nn / 128);
  wmma_gemm_bt<0><<<gO, 256, GEMM_LDS, stream>>>(sb, vt, nullptr, out, Nn, D, Nn, D, 1.0f);
}
